// msaLMMixin_20298015441144
// MI455X (gfx1250) — matrix operand order repaired and checked by reading, not compiled, not run on hardware
//
#include <hip/hip_runtime.h>
#include <hip/hip_bf16.h>

// ---------------- problem constants ----------------
constexpr int C_B = 4, C_SQ = 1024, C_SKV = 512, C_D = 2048, C_KDIM = 1024;
constexpr int C_H = 16, C_HD = 128, C_INTER = 8192, C_R = 16;
constexpr int C_MT  = C_B * C_SQ;    // 4096 query rows
constexpr int C_MKV = C_B * C_SKV;   // 2048 kv rows
constexpr float C_LS    = 2.0f;                  // lora scale 32/16
constexpr float C_EPS   = 1e-6f;
constexpr float C_SCALE = 0.08838834764831845f;  // 1/sqrt(128)

typedef __bf16 bf16_t;
typedef __attribute__((ext_vector_type(16))) __bf16 v16bf;
typedef __attribute__((ext_vector_type(8)))  __bf16 v8bf;
typedef __attribute__((ext_vector_type(8)))  float  v8f;

__device__ inline v16bf make_frag(const bf16_t* p0, const bf16_t* p1) {
  v8bf lo = *(const v8bf*)p0;
  v8bf hi = *(const v8bf*)p1;
  return __builtin_shufflevector(lo, hi, 0,1,2,3,4,5,6,7,8,9,10,11,12,13,14,15);
}
__device__ inline v8f wmma_bf16(v16bf a, v16bf b, v8f c) {
  return __builtin_amdgcn_wmma_f32_16x16x32_bf16(false, a, false, b, (short)0, c, false, false);
}

// ---- CDNA5 async global->LDS copy (ASYNCcnt path), 16B per lane ----
// LDS address = low 32 bits of the generic pointer (flat aperture keeps the
// LDS offset in addr[31:0]); global address as 64-bit VGPR pair, GV mode.
__device__ inline void async_copy_b128(const void* g, void* l) {
  unsigned lds = (unsigned)(uintptr_t)l;
  unsigned long long ga = (unsigned long long)(uintptr_t)g;
  asm volatile("global_load_async_to_lds_b128 %0, %1, off"
               :: "v"(lds), "v"(ga) : "memory");
}
__device__ inline void wait_async0() {
  asm volatile("s_wait_asynccnt 0" ::: "memory");
}

// ---------------- small helper kernels ----------------

// flat f32 -> bf16 convert
__global__ void k_cvt(const float* __restrict__ s, bf16_t* __restrict__ d, long n) {
  long i = (long)blockIdx.x * blockDim.x + threadIdx.x;
  long stride = (long)gridDim.x * blockDim.x;
  for (; i < n; i += stride) d[i] = (bf16_t)s[i];
}

// tiled transpose + convert: src f32 [Kd][Nd] -> dst bf16 [Nd][Kd], per z slice
__global__ __launch_bounds__(256) void k_tcvt(const float* __restrict__ src,
                                              bf16_t* __restrict__ dst, int Kd, int Nd) {
  __shared__ float t[32][33];
  long zoff = (long)blockIdx.z * Kd * Nd;
  src += zoff; dst += zoff;
  int n0 = blockIdx.x * 32, k0 = blockIdx.y * 32;
  int tx = threadIdx.x & 31, ty = threadIdx.x >> 5;
  #pragma unroll
  for (int j = 0; j < 4; j++) t[ty + 8 * j][tx] = src[(long)(k0 + ty + 8 * j) * Nd + n0 + tx];
  __syncthreads();
  #pragma unroll
  for (int j = 0; j < 4; j++) dst[(long)(n0 + ty + 8 * j) * Kd + k0 + tx] = (bf16_t)t[tx][ty + 8 * j];
}

// per-row rms factor of x_q, and convert raw x_q -> bf16
__global__ __launch_bounds__(256) void k_rowrs(const float* __restrict__ xq,
                                               bf16_t* __restrict__ xqb, float* __restrict__ rowrs) {
  int r = blockIdx.x;
  const float* x = xq + (long)r * C_D;
  __shared__ float red[256];
  float ss = 0.f, vals[8];
  #pragma unroll
  for (int i = 0; i < 8; i++) { float v = x[threadIdx.x + i * 256]; vals[i] = v; ss += v * v; }
  red[threadIdx.x] = ss; __syncthreads();
  for (int s = 128; s > 0; s >>= 1) { if (threadIdx.x < s) red[threadIdx.x] += red[threadIdx.x + s]; __syncthreads(); }
  if (threadIdx.x == 0) rowrs[r] = rsqrtf(red[0] * (1.0f / C_D) + C_EPS);
  #pragma unroll
  for (int i = 0; i < 8; i++) xqb[(long)r * C_D + threadIdx.x + i * 256] = (bf16_t)vals[i];
}

// q_pool[b][d] = ln1[d]/S_Q * sum_s x[b][s][d]*rowrs[b*S_Q+s]   (deterministic column reduce)
__global__ __launch_bounds__(256) void k_pool(const float* __restrict__ xq, const float* __restrict__ rowrs,
                                              const float* __restrict__ ln1, float* __restrict__ qpool) {
  int d = blockIdx.x * 256 + threadIdx.x;
  int b = blockIdx.y;
  float acc = 0.f;
  for (int s = 0; s < C_SQ; ++s)
    acc += xq[((long)b * C_SQ + s) * C_D + d] * rowrs[b * C_SQ + s];
  qpool[b * C_D + d] = ln1[d] * acc * (1.0f / C_SQ);
}

// router: logits, softmax, top-2, combine weights (with sigmoid(alpha1) folded); sig2 scalar
__global__ __launch_bounds__(256) void k_router(const float* __restrict__ qpool, const float* __restrict__ rW,
                                                const float* __restrict__ rb, const float* __restrict__ a1,
                                                const float* __restrict__ a2, float* __restrict__ coef,
                                                float* __restrict__ scalars) {
  __shared__ float logits[16];
  int tid = threadIdx.x;
  int pair = tid >> 4;          // 16 (b,e) pairs
  int b = pair >> 2, e = pair & 3;
  int sub = tid & 15;
  float s = 0.f;
  for (int k = sub; k < C_D; k += 16) s += qpool[b * C_D + k] * rW[k * 4 + e];
  for (int off = 8; off > 0; off >>= 1) s += __shfl_xor(s, off, 16);
  if (sub == 0) logits[pair] = s + rb[e];
  __syncthreads();
  if (tid < C_B) {
    int b2 = tid;
    float l[4]; float mx = -1e30f;
    for (int e2 = 0; e2 < 4; e2++) { l[e2] = logits[b2 * 4 + e2]; mx = fmaxf(mx, l[e2]); }
    float sum = 0.f;
    for (int e2 = 0; e2 < 4; e2++) { l[e2] = __expf(l[e2] - mx); sum += l[e2]; }
    for (int e2 = 0; e2 < 4; e2++) l[e2] /= sum;
    int i1 = 0; for (int e2 = 1; e2 < 4; e2++) if (l[e2] > l[i1]) i1 = e2;
    int i2 = -1; for (int e2 = 0; e2 < 4; e2++) if (e2 != i1 && (i2 < 0 || l[e2] > l[i2])) i2 = e2;
    float denom = l[i1] + l[i2] + 1e-10f;
    float sig1 = 1.f / (1.f + __expf(-a1[0]));
    for (int e2 = 0; e2 < 4; e2++) {
      float w = (e2 == i1 || e2 == i2) ? l[e2] / denom : 0.f;
      coef[b2 * 4 + e2] = sig1 * w;
    }
  }
  if (tid == 0) scalars[0] = 1.f / (1.f + __expf(-a2[0]));
}

// rmsnorm(x_f)*ln2 -> bf16 h
__global__ __launch_bounds__(256) void k_rmsnorm_h(const float* __restrict__ xf, const float* __restrict__ ln2,
                                                   bf16_t* __restrict__ hb) {
  int r = blockIdx.x;
  const float* x = xf + (long)r * C_D;
  __shared__ float red[256];
  float ss = 0.f, vals[8];
  #pragma unroll
  for (int i = 0; i < 8; i++) { float v = x[threadIdx.x + i * 256]; vals[i] = v; ss += v * v; }
  red[threadIdx.x] = ss; __syncthreads();
  for (int s = 128; s > 0; s >>= 1) { if (threadIdx.x < s) red[threadIdx.x] += red[threadIdx.x + s]; __syncthreads(); }
  float rs = rsqrtf(red[0] * (1.0f / C_D) + C_EPS);
  #pragma unroll
  for (int i = 0; i < 8; i++) {
    int d = threadIdx.x + i * 256;
    hb[(long)r * C_D + d] = (bf16_t)(ln2[d] * vals[i] * rs);
  }
}

// skinny LoRA-A: out[M][16] = X_bf16[M][K] @ A_f32[K][16]
__global__ __launch_bounds__(256) void k_lora_a(const bf16_t* __restrict__ X, const float* __restrict__ Aw,
                                                float* __restrict__ out, int K) {
  int r = blockIdx.x;
  int j = threadIdx.x & 15, part = threadIdx.x >> 4;
  float s = 0.f;
  for (int k = part; k < K; k += 16) s += (float)X[(long)r * K + k] * Aw[k * C_R + j];
  __shared__ float red[256];
  red[threadIdx.x] = s; __syncthreads();
  if (part == 0) {
    float tot = 0.f;
    #pragma unroll
    for (int p = 0; p < 16; p++) tot += red[p * 16 + j];
    out[r * C_R + j] = tot;
  }
}

// ---------------- generic WMMA GEMM (double-buffered async-LDS staging) ----------------
// C[M][N] = A_bf16[M][K] @ BT_bf16[N][K]^T  with templated epilogues.
enum { M_BIAS_BF16 = 0, M_WO = 1, M_UP = 2, M_GATE = 3, M_DOWN = 4 };
constexpr int GM = 128, GN = 128, GK = 32, SA = 40;  // SA: padded LDS k-stride (bf16 elems)

template <int MODE>
__global__ __launch_bounds__(256)
void k_gemm(const bf16_t* __restrict__ Abase, const bf16_t* __restrict__ BTbase,
            int M, int N, int K,
            long sAz, long sBz, long sOz,
            const float* __restrict__ biasBase, int biasZ,
            bf16_t* __restrict__ outBF, float* __restrict__ outF,
            const float* __restrict__ addF,
            const float* __restrict__ rowA, const float* __restrict__ loraB,
            const float* __restrict__ coef, const float* __restrict__ scalars, int eIdx) {
  __shared__ bf16_t As[2][GM * SA];
  __shared__ bf16_t Bs[2][GN * SA];
  const int tid = threadIdx.x;
  const int z = blockIdx.z;
  const bf16_t* A  = Abase  + (long)z * sAz;
  const bf16_t* BT = BTbase + (long)z * sBz;
  const int row0 = blockIdx.y * GM;
  const int col0 = blockIdx.x * GN;
  const int wid = tid >> 5, lane = tid & 31;
  const int wm = wid & 1, wn = wid >> 1;
  const int khalf = lane >> 4, l15 = lane & 15;

  // async staging of one 128x32 k-tile per operand: 2 x b128 per thread each
  auto stage = [&](int buf, int k0) {
    #pragma unroll
    for (int it = 0; it < 2; ++it) {
      int c = tid + it * 256;           // 512 16B chunks per operand
      int r = c >> 2, kc = (c & 3) * 8;
      async_copy_b128(&A [(long)(row0 + r) * K + k0 + kc], &As[buf][r * SA + kc]);
      async_copy_b128(&BT[(long)(col0 + r) * K + k0 + kc], &Bs[buf][r * SA + kc]);
    }
  };

  v8f acc[4][2] = {};
  stage(0, 0);
  wait_async0();
  __syncthreads();
  int buf = 0;
  for (int k0 = 0; k0 < K; k0 += GK) {
    if (k0 + GK < K) stage(buf ^ 1, k0 + GK);     // overlap next tile with WMMA
    if (k0 + 2 * GK < K) {                        // L2 prefetch two tiles ahead
      __builtin_prefetch(&A [(long)(row0 + (tid >> 1)) * K + k0 + 2 * GK], 0, 1);
      __builtin_prefetch(&BT[(long)(col0 + (tid >> 1)) * K + k0 + 2 * GK], 0, 1);
    }
    v16bf af[4], bfr[2];
    #pragma unroll
    for (int tm = 0; tm < 4; tm++) {
      int r = wm * 64 + tm * 16 + l15;
      af[tm] = make_frag(&As[buf][r * SA + khalf * 8], &As[buf][r * SA + khalf * 8 + 16]);
    }
    #pragma unroll
    for (int tn = 0; tn < 2; tn++) {
      int n = wn * 32 + tn * 16 + l15;
      bfr[tn] = make_frag(&Bs[buf][n * SA + khalf * 8], &Bs[buf][n * SA + khalf * 8 + 16]);
    }
    #pragma unroll
    for (int tm = 0; tm < 4; tm++)
      #pragma unroll
      for (int tn = 0; tn < 2; tn++)
        acc[tm][tn] = wmma_bf16(af[tm], bfr[tn], acc[tm][tn]);
    wait_async0();          // my async writes to buf^1 have landed in LDS
    __syncthreads();        // everyone's writes visible; all reads of buf done
    buf ^= 1;
  }

  const float* bias = biasBase ? biasBase + (long)z * biasZ : nullptr;
  #pragma unroll
  for (int tm = 0; tm < 4; tm++) {
    #pragma unroll
    for (int tn = 0; tn < 2; tn++) {
      #pragma unroll
      for (int i = 0; i < 8; i++) {
        int r = row0 + wm * 64 + tm * 16 + i + 8 * khalf;
        int c = col0 + wn * 32 + tn * 16 + l15;
        float v = acc[tm][tn][i];
        float lora = 0.f;
        if (MODE == M_UP || MODE == M_GATE || MODE == M_DOWN) {
          #pragma unroll
          for (int j = 0; j < C_R; j++) lora += rowA[(long)r * C_R + j] * loraB[(long)j * N + c];
          lora *= C_LS;
        }
        if (MODE == M_BIAS_BF16) {
          outBF[(long)z * sOz + (long)r * N + c] = (bf16_t)(v + bias[c]);
        } else if (MODE == M_WO) {
          float s = coef[((r >> 10) << 2) + eIdx];      // r>>10: batch index (S_Q=1024)
          float* p = &outF[(long)r * N + c];
          *p = *p + s * (v + bias[c]);
        } else if (MODE == M_UP) {
          outF[(long)r * N + c] = v + lora;
        } else if (MODE == M_GATE) {
          float pre = v + lora;
          float g = pre / (1.f + __expf(-pre));         // silu
          outBF[(long)r * N + c] = (bf16_t)(g + addF[(long)r * N + c]);
        } else if (MODE == M_DOWN) {
          outF[(long)r * N + c] = addF[(long)r * N + c] + scalars[0] * (v + lora);
        }
      }
    }
  }
}

// ---------------- fused cross-attention (per expert/b/head/64-row q tile) ----------------
constexpr int AT_M = 64;
constexpr int SQs = 136;   // Q/K LDS stride (bf16 elems, 128+8 pad)
constexpr int SPs = 520;   // P / V^T LDS stride (512+8 pad)
constexpr int ATT_LDS = 17408 /*Qs*/ + 148480 /*Ks+Sf | Vt*/ + 66560 /*Ps*/;  // 232448 B

__global__ __launch_bounds__(256)
void k_attn(const bf16_t* __restrict__ Qb, const bf16_t* __restrict__ Kb,
            const bf16_t* __restrict__ Vb, bf16_t* __restrict__ Ob) {
  extern __shared__ char smem[];
  bf16_t* Qs = (bf16_t*)smem;                            // 64 x 136
  char*   u  = smem + 17408;
  bf16_t* Ks = (bf16_t*)u;                               // 64 x 136
  float*  Sf = (float*)(u + 17408);                      // 64 x 512 f32
  bf16_t* Vt = (bf16_t*)u;                               // 128 x 520 (reuses Ks+Sf region)
  bf16_t* Ps = (bf16_t*)(smem + 17408 + 148480);         // 64 x 520

  const int tid = threadIdx.x, lane = tid & 31, wid = tid >> 5;
  const int e = blockIdx.z >> 2, b = blockIdx.z & 3, hh = blockIdx.y;
  const int m0 = blockIdx.x * AT_M;
  const bf16_t* Qp = Qb + ((long)e * C_MT + b * C_SQ + m0) * C_D + hh * C_HD;
  const bf16_t* Kp = Kb + ((long)e * C_MKV + b * C_SKV) * C_D + hh * C_HD;
  const bf16_t* Vp = Vb + ((long)e * C_MKV + b * C_SKV) * C_D + hh * C_HD;
  bf16_t*       Op = Ob + ((long)e * C_MT + b * C_SQ + m0) * C_D + hh * C_HD;
  const int wm = wid & 1, wn = wid >> 1;
  const int khalf = lane >> 4, l15 = lane & 15;

  // stage Q tile 64x128 (async)
  #pragma unroll
  for (int it = 0; it < 4; ++it) {
    int c = tid + it * 256;
    int r = c >> 4, kc = (c & 15) * 8;
    async_copy_b128(&Qp[(long)r * C_D + kc], &Qs[r * SQs + kc]);
  }

  // -------- scores: S = scale * Q K^T  -> Sf[64][512] --------
  for (int kv0 = 0; kv0 < C_SKV; kv0 += 64) {
    __syncthreads();                       // all waves done reading Ks (noop on iter 0)
    #pragma unroll
    for (int it = 0; it < 4; ++it) {
      int c = tid + it * 256;
      int r = c >> 4, kc = (c & 15) * 8;
      async_copy_b128(&Kp[(long)(kv0 + r) * C_D + kc], &Ks[r * SQs + kc]);
    }
    wait_async0();                         // covers Q staging too on iter 0
    __syncthreads();
    v8f acc[2] = {};
    #pragma unroll
    for (int kt = 0; kt < 4; kt++) {
      int n = wn * 16 + l15, kb_ = kt * 32 + khalf * 8;
      v16bf bfr = make_frag(&Ks[n * SQs + kb_], &Ks[n * SQs + kb_ + 16]);
      #pragma unroll
      for (int tm = 0; tm < 2; tm++) {
        int r = wm * 32 + tm * 16 + l15;
        int ka = kt * 32 + khalf * 8;
        v16bf af = make_frag(&Qs[r * SQs + ka], &Qs[r * SQs + ka + 16]);
        acc[tm] = wmma_bf16(af, bfr, acc[tm]);
      }
    }
    #pragma unroll
    for (int tm = 0; tm < 2; tm++)
      #pragma unroll
      for (int i = 0; i < 8; i++) {
        int r = wm * 32 + tm * 16 + i + 8 * khalf;
        int c = kv0 + wn * 16 + l15;
        Sf[r * C_SKV + c] = acc[tm][i] * C_SCALE;
      }
  }
  __syncthreads();

  // -------- softmax over 512 (4 threads / row) -> Ps bf16 --------
  {
    int r = tid >> 2, q4 = tid & 3;
    float mx = -1e30f;
    for (int j = q4 * 128; j < q4 * 128 + 128; ++j) mx = fmaxf(mx, Sf[r * C_SKV + j]);
    for (int off = 1; off < 4; off <<= 1) mx = fmaxf(mx, __shfl_xor(mx, off, 4));
    float sum = 0.f;
    for (int j = q4 * 128; j < q4 * 128 + 128; ++j) {
      float ev = __expf(Sf[r * C_SKV + j] - mx);
      Sf[r * C_SKV + j] = ev; sum += ev;
    }
    for (int off = 1; off < 4; off <<= 1) sum += __shfl_xor(sum, off, 4);
    float inv = 1.f / sum;
    for (int j = q4 * 128; j < q4 * 128 + 128; ++j) Ps[r * SPs + j] = (bf16_t)(Sf[r * C_SKV + j] * inv);
  }
  __syncthreads();

  // -------- stage V^T [128 hd][512 kv] (overwrites Ks/Sf region) --------
  for (int it = 0; it < 32; ++it) {
    int c = tid + it * 256;               // 8192 chunks of 8
    int kv = c >> 4, hc = (c & 15) * 8;
    v8bf vv = *(const v8bf*)&Vp[(long)kv * C_D + hc];
    #pragma unroll
    for (int x = 0; x < 8; x++) Vt[(hc + x) * SPs + kv] = vv[x];
  }
  __syncthreads();

  // -------- O = P @ V  (64 x 128) --------
  {
    v8f acc[2][2] = {};
    for (int kt = 0; kt < 16; kt++) {
      v16bf afr[2], bfr[2];
      #pragma unroll
      for (int tm = 0; tm < 2; tm++) {
        int r = wm * 32 + tm * 16 + l15, ka = kt * 32 + khalf * 8;
        afr[tm] = make_frag(&Ps[r * SPs + ka], &Ps[r * SPs + ka + 16]);
      }
      #pragma unroll
      for (int tn = 0; tn < 2; tn++) {
        int n = wn * 32 + tn * 16 + l15, kb_ = kt * 32 + khalf * 8;
        bfr[tn] = make_frag(&Vt[n * SPs + kb_], &Vt[n * SPs + kb_ + 16]);
      }
      #pragma unroll
      for (int tm = 0; tm < 2; tm++)
        #pragma unroll
        for (int tn = 0; tn < 2; tn++)
          acc[tm][tn] = wmma_bf16(afr[tm], bfr[tn], acc[tm][tn]);
    }
    #pragma unroll
    for (int tm = 0; tm < 2; tm++)
      #pragma unroll
      for (int tn = 0; tn < 2; tn++)
        #pragma unroll
        for (int i = 0; i < 8; i++) {
          int r = wm * 32 + tm * 16 + i + 8 * khalf;
          int c = wn * 32 + tn * 16 + l15;
          Op[(long)r * C_D + c] = (bf16_t)acc[tm][tn][i];
        }
  }
}

// ---------------- host side ----------------
extern "C" void kernel_launch(void* const* d_in, const int* in_sizes, int n_in,
                              void* d_out, int out_size, void* d_ws, size_t ws_size,
                              hipStream_t stream) {
  const float* x_q = (const float*)d_in[0];
  const float* z_a = (const float*)d_in[1];
  const float* z_v = (const float*)d_in[2];
  const float* z_av= (const float*)d_in[3];
  const float* ln1 = (const float*)d_in[4];
  const float* ln2 = (const float*)d_in[5];
  const float* rW  = (const float*)d_in[6];
  const float* rb  = (const float*)d_in[7];
  const float* Wq  = (const float*)d_in[8];
  const float* bq  = (const float*)d_in[9];
  const float* Wk  = (const float*)d_in[10];
  const float* bk  = (const float*)d_in[11];
  const float* Wv  = (const float*)d_in[12];
  const float* bv  = (const float*)d_in[13];
  const float* Wo  = (const float*)d_in[14];
  const float* bo  = (const float*)d_in[15];
  const float* gW  = (const float*)d_in[16];
  const float* uW  = (const float*)d_in[17];
  const float* dW  = (const float*)d_in[18];
  const float* lgA = (const float*)d_in[19];
  const float* lgB = (const float*)d_in[20];
  const float* luA = (const float*)d_in[21];
  const float* luB = (const float*)d_in[22];
  const float* ldA = (const float*)d_in[23];
  const float* ldB = (const float*)d_in[24];
  const float* a1  = (const float*)d_in[25];
  const float* a2  = (const float*)d_in[26];
  float* out = (float*)d_out;

  // -------- workspace carve (all buffers re-written every call) --------
  char* w = (char*)d_ws;
  size_t off = 0;
  auto alloc = [&](size_t bytes) -> void* {
    void* p = w + off;
    off = (off + bytes + 255) & ~(size_t)255;
    return p;
  };
  bf16_t* xq_bf = (bf16_t*)alloc((size_t)C_MT * C_D * 2);
  bf16_t* z_bf  = (bf16_t*)alloc((size_t)3 * C_MKV * C_KDIM * 2);
  bf16_t* WqT   = (bf16_t*)alloc((size_t)3 * C_D * C_D * 2);
  bf16_t* WkT   = (bf16_t*)alloc((size_t)3 * C_D * C_KDIM * 2);
  bf16_t* WvT   = (bf16_t*)alloc((size_t)3 * C_D * C_KDIM * 2);
  bf16_t* WoT   = (bf16_t*)alloc((size_t)3 * C_D * C_D * 2);
  bf16_t* gT    = (bf16_t*)alloc((size_t)C_INTER * C_D * 2);
  bf16_t* uT    = (bf16_t*)alloc((size_t)C_INTER * C_D * 2);
  bf16_t* dT    = (bf16_t*)alloc((size_t)C_D * C_INTER * 2);
  bf16_t* Qb    = (bf16_t*)alloc((size_t)3 * C_MT * C_D * 2);
  bf16_t* Kb    = (bf16_t*)alloc((size_t)3 * C_MKV * C_D * 2);
  bf16_t* Vb    = (bf16_t*)alloc((size_t)3 * C_MKV * C_D * 2);
  bf16_t* Ob    = (bf16_t*)alloc((size_t)3 * C_MT * C_D * 2);
  bf16_t* h_bf  = (bf16_t*)alloc((size_t)C_MT * C_D * 2);
  bf16_t* din_bf= (bf16_t*)alloc((size_t)C_MT * C_INTER * 2);
  float* rowrs  = (float*)alloc((size_t)C_MT * 4);
  float* qpool  = (float*)alloc((size_t)C_B * C_D * 4);
  float* coef   = (float*)alloc(16 * 4);
  float* scal   = (float*)alloc(4 * 4);
  float* x_f    = (float*)alloc((size_t)C_MT * C_D * 4);
  float* Ubuf   = (float*)alloc((size_t)C_MT * C_INTER * 4);
  float* hA_g   = (float*)alloc((size_t)C_MT * C_R * 4);
  float* hA_u   = (float*)alloc((size_t)C_MT * C_R * 4);
  float* dA     = (float*)alloc((size_t)C_MT * C_R * 4);
  (void)ws_size; (void)n_in; (void)in_sizes; (void)out_size;

  // -------- router path + activation converts --------
  k_rowrs<<<C_MT, 256, 0, stream>>>(x_q, xq_bf, rowrs);
  k_pool<<<dim3(C_D / 256, C_B), 256, 0, stream>>>(x_q, rowrs, ln1, qpool);
  k_router<<<1, 256, 0, stream>>>(qpool, rW, rb, a1, a2, coef, scal);

  long nz = (long)C_MKV * C_KDIM;
  k_cvt<<<2048, 256, 0, stream>>>(z_a,  z_bf + 0 * nz, nz);
  k_cvt<<<2048, 256, 0, stream>>>(z_v,  z_bf + 1 * nz, nz);
  k_cvt<<<2048, 256, 0, stream>>>(z_av, z_bf + 2 * nz, nz);

  // -------- weight transpose-converts: [K][N] f32 -> [N][K] bf16 --------
  k_tcvt<<<dim3(C_D / 32, C_D / 32, 3), 256, 0, stream>>>(Wq, WqT, C_D, C_D);
  k_tcvt<<<dim3(C_D / 32, C_KDIM / 32, 3), 256, 0, stream>>>(Wk, WkT, C_KDIM, C_D);
  k_tcvt<<<dim3(C_D / 32, C_KDIM / 32, 3), 256, 0, stream>>>(Wv, WvT, C_KDIM, C_D);
  k_tcvt<<<dim3(C_D / 32, C_D / 32, 3), 256, 0, stream>>>(Wo, WoT, C_D, C_D);
  k_tcvt<<<dim3(C_INTER / 32, C_D / 32, 1), 256, 0, stream>>>(gW, gT, C_D, C_INTER);
  k_tcvt<<<dim3(C_INTER / 32, C_D / 32, 1), 256, 0, stream>>>(uW, uT, C_D, C_INTER);
  k_tcvt<<<dim3(C_D / 32, C_INTER / 32, 1), 256, 0, stream>>>(dW, dT, C_INTER, C_D);

  // -------- QKV projections (batched over 3 experts) --------
  k_gemm<M_BIAS_BF16><<<dim3(C_D / GN, C_MT / GM, 3), 256, 0, stream>>>(
      xq_bf, WqT, C_MT, C_D, C_D, 0L, (long)C_D * C_D, (long)C_MT * C_D,
      bq, C_D, Qb, nullptr, nullptr, nullptr, nullptr, nullptr, nullptr, 0);
  k_gemm<M_BIAS_BF16><<<dim3(C_D / GN, C_MKV / GM, 3), 256, 0, stream>>>(
      z_bf, WkT, C_MKV, C_D, C_KDIM, (long)C_MKV * C_KDIM, (long)C_D * C_KDIM, (long)C_MKV * C_D,
      bk, C_D, Kb, nullptr, nullptr, nullptr, nullptr, nullptr, nullptr, 0);
  k_gemm<M_BIAS_BF16><<<dim3(C_D / GN, C_MKV / GM, 3), 256, 0, stream>>>(
      z_bf, WvT, C_MKV, C_D, C_KDIM, (long)C_MKV * C_KDIM, (long)C_D * C_KDIM, (long)C_MKV * C_D,
      bv, C_D, Vb, nullptr, nullptr, nullptr, nullptr, nullptr, nullptr, 0);

  // -------- fused attention --------
  hipFuncSetAttribute(reinterpret_cast<const void*>(&k_attn),
                      hipFuncAttributeMaxDynamicSharedMemorySize, ATT_LDS);
  k_attn<<<dim3(C_SQ / AT_M, C_H, 3 * C_B), 256, ATT_LDS, stream>>>(Qb, Kb, Vb, Ob);

  // -------- x_f = x_q + sum_e coef * (O_e Wo_e + bo_e)  (sequential, no races) --------
  hipMemcpyAsync(x_f, x_q, (size_t)C_MT * C_D * 4, hipMemcpyDeviceToDevice, stream);
  for (int e = 0; e < 3; ++e) {
    k_gemm<M_WO><<<dim3(C_D / GN, C_MT / GM, 1), 256, 0, stream>>>(
        Ob + (long)e * C_MT * C_D, WoT + (long)e * C_D * C_D, C_MT, C_D, C_D,
        0L, 0L, 0L, bo + (long)e * C_D, 0, nullptr, x_f, nullptr, nullptr, nullptr, coef, nullptr, e);
  }

  // -------- LoRA Llama MLP --------
  k_rmsnorm_h<<<C_MT, 256, 0, stream>>>(x_f, ln2, h_bf);
  k_lora_a<<<C_MT, 256, 0, stream>>>(h_bf, lgA, hA_g, C_D);
  k_lora_a<<<C_MT, 256, 0, stream>>>(h_bf, luA, hA_u, C_D);

  k_gemm<M_UP><<<dim3(C_INTER / GN, C_MT / GM, 1), 256, 0, stream>>>(
      h_bf, uT, C_MT, C_INTER, C_D, 0L, 0L, 0L, nullptr, 0,
      nullptr, Ubuf, nullptr, hA_u, luB, nullptr, nullptr, 0);
  k_gemm<M_GATE><<<dim3(C_INTER / GN, C_MT / GM, 1), 256, 0, stream>>>(
      h_bf, gT, C_MT, C_INTER, C_D, 0L, 0L, 0L, nullptr, 0,
      din_bf, nullptr, Ubuf, hA_g, lgB, nullptr, nullptr, 0);

  k_lora_a<<<C_MT, 256, 0, stream>>>(din_bf, ldA, dA, C_INTER);
  k_gemm<M_DOWN><<<dim3(C_D / GN, C_MT / GM, 1), 256, 0, stream>>>(
      din_bf, dT, C_MT, C_D, C_INTER, 0L, 0L, 0L, nullptr, 0,
      nullptr, out, x_f, dA, ldB, nullptr, scal, 0);
}
